// MambaEncoderDecoder_78125455114997
// MI455X (gfx1250) — matrix operand order repaired and checked by reading, not compiled, not run on hardware
//
#include <hip/hip_runtime.h>
#include <hip/hip_bf16.h>

// ---------------------------------------------------------------------------
// Mamba encoder/decoder for MI455X (gfx1250, wave32, WMMA).
// Dense projections: v_wmma_f32_16x16x32_f16, 2x2 tiles per wave (32x32),
// block tile 64x128, double-buffered LDS (one barrier per K-step), weights
// pre-converted to f16 [N][K] (zero-padded to N%128==0). B tiles use async
// global->LDS loads (ASYNCcnt) when the toolchain exposes the builtin.
// Requires: M % 64 == 0, K % 32 == 0 (true for every GEMM here).
// Scan/conv/LN/mix are fused VALU kernels; ~90MB working set lives in L2.
// ---------------------------------------------------------------------------

typedef __attribute__((ext_vector_type(16))) _Float16 v16h;
typedef __attribute__((ext_vector_type(8)))  _Float16 v8h;
typedef __attribute__((ext_vector_type(8)))  float    v8f;
typedef __attribute__((ext_vector_type(4)))  int      v4i;

#if defined(__has_builtin)
#if __has_builtin(__builtin_amdgcn_global_load_async_to_lds_b128) && \
    __has_builtin(__builtin_amdgcn_s_wait_asynccnt)
#define USE_ASYNC_LDS 1
#endif
#endif
#ifndef USE_ASYNC_LDS
#define USE_ASYNC_LDS 0
#endif

#define TM 64     // block tile M
#define TN 128    // block tile N
#define TK 32     // K step
#define LDSPAD 40 // padded LDS row stride (halves): 80B = 16B-aligned, bank-spread

union V16u { v16h v; v8h h[2]; };

// C[M,N] = A[M,K] @ BT[N][K]^T (+bias).  A f32 row-major, BT f16 [Npad][K].
// M%64==0, K%32==0, grid.x covers Npad/128 (BT zero-padded), stores guarded by N.
__global__ __launch_bounds__(256)
void wmma_gemm_k(const float* __restrict__ A, const _Float16* __restrict__ BT,
                 const float* __restrict__ bias, float* __restrict__ C,
                 int M, int N, int K)
{
    __shared__ _Float16 sA [2][TM][LDSPAD];   // [buf][m][k]
    __shared__ _Float16 sBT[2][TN][LDSPAD];   // [buf][n][k]

    const int tid  = threadIdx.x;
    const int lane = tid & 31;
    const int wave = tid >> 5;       // 8 waves
    const int wm2  = wave >> 2;      // 0..1 : 32-row band
    const int wn2  = wave & 3;       // 0..3 : 32-col band
    const int m0   = blockIdx.y * TM;
    const int n0   = blockIdx.x * TN;

    const int hl  = lane & 15;
    const int hi  = lane >> 4;
    const int kbA = hi * 16;         // A half-wave K base
    const int kbB = hi * 16;         // B half-wave K base

    // staging-thread coordinates (fixed per thread)
    const int arow = tid >> 2;             // 0..63   (A: 64 rows x 4 8-half chunks)
    const int aq   = (tid & 3) * 8;        // halves offset 0/8/16/24
    const int brow = tid >> 1;             // 0..127  (B: 128 rows x 2 16-half chunks)
    const int bq   = (tid & 1) * 16;       // halves offset 0/16

    const float*    aBase = &A [(size_t)(m0 + arow) * K + aq];
    const _Float16* bBase = &BT[(size_t)(n0 + brow) * K + bq];

    v8f acc00 = {0.f,0.f,0.f,0.f,0.f,0.f,0.f,0.f};
    v8f acc01 = acc00, acc10 = acc00, acc11 = acc00;

    float4 ra0, ra1;                 // A register staging (needs f32->f16 cvt)
#if !USE_ASYNC_LDS
    v8h rb0, rb1;                    // B register staging (sync fallback)
#endif

    // helpers -----------------------------------------------------------
    auto fetchA = [&](int k) {
        ra0 = *(const float4*)(aBase + k);
        ra1 = *(const float4*)(aBase + k + 4);
    };
    auto storeA = [&](int buf) {
        v8h hv;
        hv[0] = (_Float16)ra0.x; hv[1] = (_Float16)ra0.y;
        hv[2] = (_Float16)ra0.z; hv[3] = (_Float16)ra0.w;
        hv[4] = (_Float16)ra1.x; hv[5] = (_Float16)ra1.y;
        hv[6] = (_Float16)ra1.z; hv[7] = (_Float16)ra1.w;
        *(v8h*)&sA[buf][arow][aq] = hv;
    };
#if USE_ASYNC_LDS
    auto copyB = [&](int k, int buf) {   // async global->LDS, 2 x b128 per thread
        const _Float16* gp = bBase + k;
        __builtin_amdgcn_global_load_async_to_lds_b128(
            (__attribute__((address_space(1))) v4i*)gp,
            (__attribute__((address_space(3))) v4i*)&sBT[buf][brow][bq], 0, 0);
        __builtin_amdgcn_global_load_async_to_lds_b128(
            (__attribute__((address_space(1))) v4i*)(gp + 8),
            (__attribute__((address_space(3))) v4i*)&sBT[buf][brow][bq + 8], 0, 0);
    };
#else
    auto fetchB = [&](int k) {
        rb0 = *(const v8h*)(bBase + k);
        rb1 = *(const v8h*)(bBase + k + 8);
    };
    auto storeB = [&](int buf) {
        *(v8h*)&sBT[buf][brow][bq]     = rb0;
        *(v8h*)&sBT[buf][brow][bq + 8] = rb1;
    };
#endif

    // ---- prologue: tile 0 into buffer 0 ----
    fetchA(0);
    storeA(0);
#if USE_ASYNC_LDS
    copyB(0, 0);
    __builtin_amdgcn_s_wait_asynccnt(0);
#else
    fetchB(0);
    storeB(0);
#endif
    __syncthreads();

    int p = 0;
    for (int k0 = 0; k0 < K; k0 += TK) {
        const int  kn    = k0 + TK;
        const bool more  = (kn < K);

        // prefetch next tile while WMMAs run
        if (more) {
            fetchA(kn);
#if USE_ASYNC_LDS
            copyB(kn, p ^ 1);
#else
            fetchB(kn);
#endif
        }

        // ---- fragments from buffer p: contiguous ds_load_b128 pairs ----
        // A 16x32 layout: element i -> k=kbA+i (contiguous 16 halves)
        const int am = wm2 * 32 + hl;
        V16u af0, af1;
        af0.h[0] = *(const v8h*)&sA[p][am][kbA];
        af0.h[1] = *(const v8h*)&sA[p][am][kbA + 8];
        af1.h[0] = *(const v8h*)&sA[p][am + 16][kbA];
        af1.h[1] = *(const v8h*)&sA[p][am + 16][kbA + 8];
        // B 32x16 layout: element i -> k=kbB+i (contiguous 16 halves)
        const int bn = wn2 * 32 + hl;
        V16u bf0, bf1;
        bf0.h[0] = *(const v8h*)&sBT[p][bn][kbB];
        bf0.h[1] = *(const v8h*)&sBT[p][bn][kbB + 8];
        bf1.h[0] = *(const v8h*)&sBT[p][bn + 16][kbB];
        bf1.h[1] = *(const v8h*)&sBT[p][bn + 16][kbB + 8];

        acc00 = __builtin_amdgcn_wmma_f32_16x16x32_f16(false, af0.v, false, bf0.v,
                                                       (short)0, acc00, false, false);
        acc01 = __builtin_amdgcn_wmma_f32_16x16x32_f16(false, af0.v, false, bf1.v,
                                                       (short)0, acc01, false, false);
        acc10 = __builtin_amdgcn_wmma_f32_16x16x32_f16(false, af1.v, false, bf0.v,
                                                       (short)0, acc10, false, false);
        acc11 = __builtin_amdgcn_wmma_f32_16x16x32_f16(false, af1.v, false, bf1.v,
                                                       (short)0, acc11, false, false);

        // stage next tile into the other buffer
        if (more) {
            storeA(p ^ 1);
#if !USE_ASYNC_LDS
            storeB(p ^ 1);
#endif
        }
#if USE_ASYNC_LDS
        __builtin_amdgcn_s_wait_asynccnt(0);
#endif
        __syncthreads();
        p ^= 1;
    }

    // ---- store: C/D layout lane n = hl, VGPR r -> M = r + 8*hi ----
    const int mA = m0 + wm2 * 32 + hi * 8;
    #pragma unroll
    for (int j = 0; j < 2; ++j) {
        int n = n0 + wn2 * 32 + j * 16 + hl;
        if (n < N) {
            float bv = bias ? bias[n] : 0.f;
            const v8f* a0 = (j == 0) ? &acc00 : &acc01;
            const v8f* a1 = (j == 0) ? &acc10 : &acc11;
            #pragma unroll
            for (int r = 0; r < 8; ++r)
                C[(size_t)(mA + r) * N + n] = (*a0)[r] + bv;
            #pragma unroll
            for (int r = 0; r < 8; ++r)
                C[(size_t)(mA + 16 + r) * N + n] = (*a1)[r] + bv;
        }
    }
}

// BT[n*K + k] = f16(B[k*N + n]) for n<N else 0  (weight transpose+convert, tiny)
__global__ void wconv_k(const float* __restrict__ B, _Float16* __restrict__ BT,
                        int N, int K, int Npad)
{
    int t = blockIdx.x * blockDim.x + threadIdx.x;
    if (t >= Npad * K) return;
    int n = t / K, k = t % K;
    BT[t] = (n < N) ? (_Float16)B[(size_t)k * N + n] : (_Float16)0.f;
}

// x[r, d] = reads[r, 0:4] @ exp_w[4,128] + exp_b  (K=4: not worth WMMA)
__global__ void expand_k(const float* __restrict__ reads,
                         const float* __restrict__ w, const float* __restrict__ b,
                         float* __restrict__ x, int rows)
{
    int t = blockIdx.x * blockDim.x + threadIdx.x;
    if (t >= rows * 128) return;
    int d = t & 127, r = t >> 7;
    float a = b[d];
    #pragma unroll
    for (int k = 0; k < 4; ++k) a += reads[r * 4 + k] * w[k * 128 + d];
    x[t] = a;
}

// u = silu(causal_conv(xi) + conv_b); xi = xz[..., :d_inner] (stride xzs)
__global__ void conv_silu_k(const float* __restrict__ xz,
                            const float* __restrict__ cw, const float* __restrict__ cb,
                            float* __restrict__ u,
                            int Bs, int L, int di, int xzs, int dconv)
{
    int t = blockIdx.x * blockDim.x + threadIdx.x;
    if (t >= Bs * L * di) return;
    int d = t % di;
    int l = (t / di) % L;
    int b = t / (di * L);
    float a = cb[d];
    for (int k = 0; k < dconv; ++k) {
        int ls = l - (dconv - 1) + k;
        if (ls >= 0) a += xz[(size_t)(b * L + ls) * xzs + d] * cw[k * di + d];
    }
    u[t] = a / (1.f + __expf(-a));   // silu
}

// Selective scan: one thread owns one (batch, channel) recurrence.
__global__ void scan_k(const float* __restrict__ u, const float* __restrict__ xdbl,
                       const float* __restrict__ xz,
                       const float* __restrict__ dtw, const float* __restrict__ dtb,
                       const float* __restrict__ Alog, const float* __restrict__ Dp,
                       float* __restrict__ y,
                       int Bs, int L, int di, int dtr, int xds, int xzs)
{
    int t = blockIdx.x * blockDim.x + threadIdx.x;
    if (t >= Bs * di) return;
    int d = t % di;
    int b = t / di;

    float A[16], h[16], wloc[16];
    #pragma unroll
    for (int s = 0; s < 16; ++s) {
        A[s] = -__expf(Alog[d * 16 + s]);
        h[s] = 0.f;
    }
    for (int r = 0; r < dtr; ++r) wloc[r] = dtw[r * di + d];
    const float dtb_v = dtb[d];
    const float Dv    = Dp[d];

    for (int l = 0; l < L; ++l) {
        const float* xr = xdbl + (size_t)(b * L + l) * xds;
        float dtv = dtb_v;
        for (int r = 0; r < dtr; ++r) dtv += xr[r] * wloc[r];
        dtv = (dtv > 20.f) ? dtv : __logf(1.f + __expf(dtv));   // softplus

        float uv = u[(size_t)(b * L + l) * di + d];
        float du = dtv * uv;
        float yv = 0.f;
        const float* Bv = xr + dtr;
        const float* Cv = xr + dtr + 16;
        #pragma unroll
        for (int s = 0; s < 16; ++s) {
            h[s] = __expf(dtv * A[s]) * h[s] + du * Bv[s];
            yv  += h[s] * Cv[s];
        }
        yv += uv * Dv;
        float zv = xz[(size_t)(b * L + l) * xzs + di + d];
        yv *= zv / (1.f + __expf(-zv));
        y[(size_t)(b * L + l) * di + d] = yv;
    }
}

// In-place layernorm (D=128, one row per 128-thread block) + relu
__global__ void ln_relu_k(float* __restrict__ x,
                          const float* __restrict__ g, const float* __restrict__ b)
{
    __shared__ float red[128];
    const int row = blockIdx.x, t = threadIdx.x;
    float v = x[(size_t)row * 128 + t];
    red[t] = v; __syncthreads();
    for (int s = 64; s > 0; s >>= 1) { if (t < s) red[t] += red[t + s]; __syncthreads(); }
    float mu = red[0] * (1.f / 128.f);
    __syncthreads();
    float dv = v - mu;
    red[t] = dv * dv; __syncthreads();
    for (int s = 64; s > 0; s >>= 1) { if (t < s) red[t] += red[t + s]; __syncthreads(); }
    float var = red[0] * (1.f / 128.f);
    float o = dv * rsqrtf(var + 1e-5f) * g[t] + b[t];
    x[(size_t)row * 128 + t] = fmaxf(o, 0.f);
}

// enc[b, 0:128] = mean_l x[b,l,:]; enc[b, 128:256] = x[b, L-1, :]
__global__ void pool_k(const float* __restrict__ x, float* __restrict__ enc, int L)
{
    int b = blockIdx.x, t = threadIdx.x;   // 128 threads
    float s = 0.f;
    for (int l = 0; l < L; ++l) s += x[(size_t)(b * L + l) * 128 + t];
    enc[b * 256 + t]        = s * (1.f / (float)L);
    enc[b * 256 + 128 + t]  = x[(size_t)(b * L + L - 1) * 128 + t];
}

__global__ void gather_k(const float* __restrict__ enc, const int* __restrict__ idx,
                         float* __restrict__ seq, int Q, int Cc)
{
    int t = blockIdx.x * blockDim.x + threadIdx.x;
    if (t >= Q * Cc) return;
    int q = t / Cc, c = t % Cc;
    seq[t] = enc[(size_t)idx[q] * Cc + c];
}

// Per (q,k): 8 head dots -> concat overlap -> 9x9 relu MLP -> 9x9 -> mean
__global__ void final_mix_k(const float* __restrict__ q, const float* __restrict__ k,
                            const float* __restrict__ ov, const int* __restrict__ idx,
                            const float* __restrict__ m1w, const float* __restrict__ m1b,
                            const float* __restrict__ m2w, const float* __restrict__ m2b,
                            float* __restrict__ out, int Q, int Bs)
{
    int t = blockIdx.x * blockDim.x + threadIdx.x;
    if (t >= Q * Bs) return;
    int ki = t % Bs, qi = t / Bs;
    float s[9];
    #pragma unroll
    for (int h = 0; h < 8; ++h) {
        float a = 0.f;
        #pragma unroll
        for (int d = 0; d < 16; ++d)
            a += q[qi * 128 + h * 16 + d] * k[ki * 128 + h * 16 + d];
        s[h] = a;
    }
    s[8] = ov[(size_t)idx[qi] * Bs + ki];
    float s1[9];
    #pragma unroll
    for (int j = 0; j < 9; ++j) {
        float a = m1b[j];
        #pragma unroll
        for (int i = 0; i < 9; ++i) a += s[i] * m1w[i * 9 + j];
        s1[j] = fmaxf(a, 0.f);
    }
    float mean = 0.f;
    #pragma unroll
    for (int j = 0; j < 9; ++j) {
        float a = m2b[j];
        #pragma unroll
        for (int i = 0; i < 9; ++i) a += s1[i] * m2w[i * 9 + j];
        mean += a;
    }
    out[(size_t)qi * Bs + ki] = mean * (1.f / 9.f);
}

static inline void launch_gemm(const float* A, const _Float16* BT, const float* bias,
                               float* C, int M, int N, int Npad, int K, hipStream_t s)
{
    dim3 g(Npad / TN, (M + TM - 1) / TM);
    hipLaunchKernelGGL(wmma_gemm_k, g, dim3(256), 0, s, A, BT, bias, C, M, N, K);
}

static inline void launch_wconv(const float* B, _Float16* BT, int N, int K, int Npad,
                                hipStream_t s)
{
    int tot = Npad * K;
    hipLaunchKernelGGL(wconv_k, dim3((tot + 255) / 256), dim3(256), 0, s, B, BT, N, K, Npad);
}

extern "C" void kernel_launch(void* const* d_in, const int* in_sizes, int n_in,
                              void* d_out, int out_size, void* d_ws, size_t ws_size,
                              hipStream_t stream)
{
    (void)in_sizes; (void)n_in; (void)out_size; (void)ws_size;
    // ---- input pointers ----
    // top-level dict insertion order; `params` flattened as JAX tree leaves
    // (dict keys sorted alphabetically at every level, lists in order).
    const float* reads = (const float*)d_in[0];
    const int*   idx   = (const int*)  d_in[1];
    const float* ovl   = (const float*)d_in[2];
    // params.dec (mamba, keys sorted): A_log, D, conv_b, conv_w, dt_b, dt_w,
    //                                  in_proj, out_proj, x_proj
    const float* dAlog = (const float*)d_in[3];
    const float* dDp   = (const float*)d_in[4];
    const float* dcb   = (const float*)d_in[5];
    const float* dcw   = (const float*)d_in[6];
    const float* ddtb  = (const float*)d_in[7];
    const float* ddtw  = (const float*)d_in[8];
    const float* dinp  = (const float*)d_in[9];
    const float* doutp = (const float*)d_in[10];
    const float* dxp   = (const float*)d_in[11];
    // params.enc_layers[i] keys sorted: ln_b, ln_g, mamba{A_log,D,conv_b,conv_w,
    //                                   dt_b,dt_w,in_proj,out_proj,x_proj}
    // base index: 12 + 11*i
    const float* expb = (const float*)d_in[45];
    const float* expw = (const float*)d_in[46];
    const float* kb   = (const float*)d_in[47];
    const float* kw   = (const float*)d_in[48];
    const float* m1b  = (const float*)d_in[49];
    const float* m1w  = (const float*)d_in[50];
    const float* m2b  = (const float*)d_in[51];
    const float* m2w  = (const float*)d_in[52];
    const float* qb   = (const float*)d_in[53];
    const float* qw   = (const float*)d_in[54];

    // ---- workspace layout ----
    const int Bsz = 64, L = 256, DM = 128, DI = 256;
    const int Mrows = Bsz * L;                 // 16384
    float* ws = (float*)d_ws;
    size_t o = 0;
    auto alloc = [&](size_t n) { float* p = ws + o; o += n; return p; };
    float* x0    = alloc((size_t)Mrows * DM);
    float* x1    = alloc((size_t)Mrows * DM);
    float* xz    = alloc((size_t)Mrows * 2 * DI);
    float* u     = alloc((size_t)Mrows * DI);
    float* xdbl  = alloc((size_t)Mrows * 40);
    float* y     = alloc((size_t)Mrows * DI);
    float* enc   = alloc((size_t)Bsz * 256);
    float* seq   = alloc((size_t)64 * 256);
    float* xzd   = alloc((size_t)64 * 1024);
    float* ud    = alloc((size_t)64 * 512);
    float* xdbld = alloc((size_t)64 * 48);
    float* yd    = alloc((size_t)64 * 512);
    float* seqo  = alloc((size_t)64 * 256);
    float* qbuf  = alloc((size_t)64 * 128);
    float* kbuf  = alloc((size_t)64 * 128);

    // f16 transposed weight arena (after the f32 buffers)
    _Float16* wh = (_Float16*)(ws + o);
    size_t ho = 0;
    auto halloc = [&](size_t n) { _Float16* p = wh + ho; ho += n; return p; };
    _Float16* w_inp[3];  for (int i = 0; i < 3; ++i) w_inp[i]  = halloc(512  * 128);
    _Float16* w_xp[3];   for (int i = 0; i < 3; ++i) w_xp[i]   = halloc(128  * 256);
    _Float16* w_outp[3]; for (int i = 0; i < 3; ++i) w_outp[i] = halloc(128  * 256);
    _Float16* w_dinp  = halloc(1024 * 256);
    _Float16* w_dxp   = halloc(128  * 512);
    _Float16* w_doutp = halloc(256  * 512);
    _Float16* w_q     = halloc(128  * 256);
    _Float16* w_k     = halloc(128  * 256);

    // ---- 0) weight convert/transpose (once per call, ~1M elements total) ----
    for (int li = 0; li < 3; ++li) {
        const int base = 12 + 11 * li;
        launch_wconv((const float*)d_in[base + 8],  w_inp[li],  512, 128, 512, stream);
        launch_wconv((const float*)d_in[base + 10], w_xp[li],   40,  256, 128, stream);
        launch_wconv((const float*)d_in[base + 9],  w_outp[li], 128, 256, 128, stream);
    }
    launch_wconv(dinp,  w_dinp,  1024, 256, 1024, stream);
    launch_wconv(dxp,   w_dxp,   48,   512, 128,  stream);
    launch_wconv(doutp, w_doutp, 256,  512, 256,  stream);
    launch_wconv(qw,    w_q,     128,  256, 128,  stream);
    launch_wconv(kw,    w_k,     128,  256, 128,  stream);

    // ---- 1) expand 4 -> 128 ----
    hipLaunchKernelGGL(expand_k, dim3((Mrows * 128 + 255) / 256), dim3(256), 0, stream,
                       reads, expw, expb, x0, Mrows);

    // ---- 2) three encoder Mamba + LN + relu layers ----
    float* xin = x0;
    float* xot = x1;
    for (int li = 0; li < 3; ++li) {
        const int base = 12 + 11 * li;
        const float* lnb  = (const float*)d_in[base + 0];
        const float* lng  = (const float*)d_in[base + 1];
        const float* Alog = (const float*)d_in[base + 2];
        const float* Dp   = (const float*)d_in[base + 3];
        const float* cb   = (const float*)d_in[base + 4];
        const float* cw   = (const float*)d_in[base + 5];
        const float* dtb  = (const float*)d_in[base + 6];
        const float* dtw  = (const float*)d_in[base + 7];

        launch_gemm(xin, w_inp[li], nullptr, xz, Mrows, 2 * DI, 512, DM, stream);
        hipLaunchKernelGGL(conv_silu_k, dim3((Mrows * DI + 255) / 256), dim3(256), 0, stream,
                           xz, cw, cb, u, Bsz, L, DI, 2 * DI, 4);
        launch_gemm(u, w_xp[li], nullptr, xdbl, Mrows, 40, 128, DI, stream);
        hipLaunchKernelGGL(scan_k, dim3((Bsz * DI + 255) / 256), dim3(256), 0, stream,
                           u, xdbl, xz, dtw, dtb, Alog, Dp, y,
                           Bsz, L, DI, 8, 40, 2 * DI);
        launch_gemm(y, w_outp[li], nullptr, xot, Mrows, DM, 128, DI, stream);
        hipLaunchKernelGGL(ln_relu_k, dim3(Mrows), dim3(128), 0, stream, xot, lng, lnb);

        float* tmp = xin; xin = xot; xot = tmp;
    }
    // xin now holds final encoder activations

    // ---- 3) pooling + gather ----
    hipLaunchKernelGGL(pool_k, dim3(Bsz), dim3(128), 0, stream, xin, enc, L);
    hipLaunchKernelGGL(gather_k, dim3((64 * 256 + 255) / 256), dim3(256), 0, stream,
                       enc, idx, seq, 64, 256);

    // ---- 4) decoder mamba (d_model=256, d_inner=512, dt_rank=16, d_conv=1) ----
    launch_gemm(seq, w_dinp, nullptr, xzd, 64, 1024, 1024, 256, stream);
    hipLaunchKernelGGL(conv_silu_k, dim3((64 * 512 + 255) / 256), dim3(256), 0, stream,
                       xzd, dcw, dcb, ud, 1, 64, 512, 1024, 1);
    launch_gemm(ud, w_dxp, nullptr, xdbld, 64, 48, 128, 512, stream);
    hipLaunchKernelGGL(scan_k, dim3((512 + 255) / 256), dim3(256), 0, stream,
                       ud, xdbld, xzd, ddtw, ddtb, dAlog, dDp, yd,
                       1, 64, 512, 16, 48, 1024);
    launch_gemm(yd, w_doutp, nullptr, seqo, 64, 256, 256, 512, stream);

    // ---- 5) q/k projections + mixing head ----
    launch_gemm(seqo, w_q, qb, qbuf, 64, 128, 128, 256, stream);
    launch_gemm(enc,  w_k, kb, kbuf, 64, 128, 128, 256, stream);
    hipLaunchKernelGGL(final_mix_k, dim3((64 * 64 + 255) / 256), dim3(256), 0, stream,
                       qbuf, kbuf, ovl, idx, m1w, m1b, m2w, m2b,
                       (float*)d_out, 64, 64);
}
